// Attention_84490596647056
// MI455X (gfx1250) — compile-verified, matrix operand order checked by reading, not run on hardware
//
#include <hip/hip_runtime.h>
#include <hip/hip_bf16.h>

// Problem constants (match reference)
#define BB 2
#define SS 2048
#define DD 3072
#define HH 24
#define KVH 8
#define HD 128
#define NREP 3   // HH / KVH

typedef __attribute__((ext_vector_type(8)))  float  v8f;
typedef __attribute__((ext_vector_type(16))) __bf16 v16bf;

// ---------- bf16 helpers (RNE) ----------
__device__ __forceinline__ unsigned short f2bf(float f) {
    unsigned int u = __builtin_bit_cast(unsigned int, f);
    unsigned int r = u + 0x7FFFu + ((u >> 16) & 1u);
    return (unsigned short)(r >> 16);
}
__device__ __forceinline__ float bf2f(unsigned short u) {
    unsigned int b = ((unsigned int)u) << 16;
    return __builtin_bit_cast(float, b);
}

union FragU { v16bf v; uint4 q[2]; };

// Fragment load from an LDS tile with row pitch 40 ushorts (80B, 16B-aligned).
// A-layout (16x32 bf16): lane L holds row (L&15); lanes<16 hold K chunks [0..7],[16..23],
// lanes>=16 hold [8..15],[24..31].  B-layout (32x16) addressed identically with "row"
// meaning the output column, K packed the same way.
__device__ __forceinline__ v16bf lds_frag(const unsigned short* tile, int rowBase, int lane) {
    const int r  = rowBase + (lane & 15);
    const int kh = (lane >> 4) * 8;
    const unsigned short* p = tile + r * 40 + kh;
    FragU u;
    u.q[0] = *(const uint4*)(p);
    u.q[1] = *(const uint4*)(p + 16);
    return u.v;
}
// Same fragment pattern, direct from global.
__device__ __forceinline__ v16bf glb_frag(const unsigned short* p) {
    FragU u;
    u.q[0] = *(const uint4*)(p);
    u.q[1] = *(const uint4*)(p + 16);
    return u.v;
}

__device__ __forceinline__ v8f wmma_bf16(v16bf a, v16bf b, v8f c) {
    return __builtin_amdgcn_wmma_f32_16x16x32_bf16(false, a, false, b, (short)0, c, false, false);
}

// Async 16B copy global -> LDS (ASYNCcnt-tracked, bypasses VGPRs).
__device__ __forceinline__ void async_cp16(unsigned ldsAddr, const unsigned short* gptr) {
    asm volatile("global_load_async_to_lds_b128 %0, %1, off"
                 :: "v"(ldsAddr), "v"(gptr) : "memory");
}

// XOR lane swizzle within 16-lane groups via ds_swizzle (no index VGPR needed).
template <int XM>
__device__ __forceinline__ float swz_xor(float x) {
    int v = __builtin_amdgcn_ds_swizzle(__builtin_bit_cast(int, x), (XM << 10) | 0x1F);
    return __builtin_bit_cast(float, v);
}
__device__ __forceinline__ float grp_max(float x) {
    x = fmaxf(x, swz_xor<1>(x));
    x = fmaxf(x, swz_xor<2>(x));
    x = fmaxf(x, swz_xor<4>(x));
    x = fmaxf(x, swz_xor<8>(x));
    return x;
}
__device__ __forceinline__ float grp_sum(float x) {
    x += swz_xor<1>(x);
    x += swz_xor<2>(x);
    x += swz_xor<4>(x);
    x += swz_xor<8>(x);
    return x;
}

// ---------- f32 -> bf16 conversion (vectorized) ----------
__global__ __launch_bounds__(256) void cvt_f32_bf16_kernel(const float* __restrict__ in,
                                                           unsigned short* __restrict__ out,
                                                           int n4) {
    for (int i = blockIdx.x * blockDim.x + threadIdx.x; i < n4; i += gridDim.x * blockDim.x) {
        float4 f = ((const float4*)in)[i];
        uint2 pk;
        pk.x = (unsigned int)f2bf(f.x) | ((unsigned int)f2bf(f.y) << 16);
        pk.y = (unsigned int)f2bf(f.z) | ((unsigned int)f2bf(f.w) << 16);
        ((uint2*)out)[i] = pk;
    }
}

// ---------- store helpers for GEMM epilogue ----------
__device__ __forceinline__ void store_elem(float* C, size_t idx, float v)          { C[idx] = v; }
__device__ __forceinline__ void store_elem(unsigned short* C, size_t idx, float v) { C[idx] = f2bf(v); }

// ---------- Tiled WMMA GEMM: C(MxN) = A(MxK) * W(NxK)^T ----------
// 256 threads (8 waves), 128x128 C tile, wave = 4x2 grid of 16x16 WMMA tiles.
// K stepped by 32 through double-buffered LDS filled by async global->LDS copies:
// issue tile k+1, s_wait_asynccnt for tile k only, barrier, compute.
template <typename OutT>
__global__ __launch_bounds__(256) void gemm_bf16_kernel(const unsigned short* __restrict__ A,
                                                        const unsigned short* __restrict__ W,
                                                        OutT* __restrict__ C,
                                                        int M, int N, int K) {
    __shared__ unsigned short As[2][128 * 40];
    __shared__ unsigned short Bs[2][128 * 40];

    const int tid  = threadIdx.x;
    const int lane = tid & 31;
    const int wave = tid >> 5;
    const int wr   = wave >> 2;   // 0..1
    const int wc   = wave & 3;    // 0..3
    const int bm   = blockIdx.y * 128;
    const int bn   = blockIdx.x * 128;

    // cooperative copy mapping: thread t -> row t>>1 (0..127), half t&1 -> 16 bf16 (32B)
    const int r    = tid >> 1;
    const int half = tid & 1;
    const size_t aBase = (size_t)(bm + r) * K + half * 16;
    const size_t bBase = (size_t)(bn + r) * K + half * 16;
    const int ldsOff   = r * 40 + half * 16;

    const unsigned ldsA0     = (unsigned)(size_t)(&As[0][ldsOff]);
    const unsigned ldsB0     = (unsigned)(size_t)(&Bs[0][ldsOff]);
    const unsigned bufStride = (unsigned)(128 * 40 * 2);   // bytes between buffers

    auto issue = [&](int buf, int kt) {
        const unsigned short* pa = A + aBase + (size_t)kt * 32;
        const unsigned short* pb = W + bBase + (size_t)kt * 32;
        const unsigned la = ldsA0 + (unsigned)buf * bufStride;
        const unsigned lb = ldsB0 + (unsigned)buf * bufStride;
        async_cp16(la,       pa);
        async_cp16(la + 16u, pa + 8);
        async_cp16(lb,       pb);
        async_cp16(lb + 16u, pb + 8);
    };

    v8f acc[4][2];
#pragma unroll
    for (int i = 0; i < 4; ++i)
#pragma unroll
        for (int j = 0; j < 2; ++j) acc[i][j] = (v8f)(0.0f);

    const int nk = K >> 5;
    issue(0, 0);
    for (int kt = 0; kt < nk; ++kt) {
        if (kt + 1 < nk) {
            issue((kt + 1) & 1, kt + 1);                    // overlap next tile's DMA
            asm volatile("s_wait_asynccnt 4" ::: "memory"); // wait for tile kt only
        } else {
            asm volatile("s_wait_asynccnt 0" ::: "memory");
        }
        __syncthreads();

        const unsigned short* Ab = As[kt & 1];
        const unsigned short* Bb = Bs[kt & 1];
        v16bf af[4], bf[2];
#pragma unroll
        for (int i = 0; i < 4; ++i) af[i] = lds_frag(Ab, wr * 64 + i * 16, lane);
#pragma unroll
        for (int j = 0; j < 2; ++j) bf[j] = lds_frag(Bb, wc * 32 + j * 16, lane);
#pragma unroll
        for (int i = 0; i < 4; ++i)
#pragma unroll
            for (int j = 0; j < 2; ++j) acc[i][j] = wmma_bf16(af[i], bf[j], acc[i][j]);

        __syncthreads();   // protect buffer kt&1 before it is refilled next-next iter
    }

    // C/D layout: VGPR v, lane L -> row = v + 8*(L>>4), col = L&15
    const int hi = lane >> 4, lo = lane & 15;
#pragma unroll
    for (int i = 0; i < 4; ++i)
#pragma unroll
        for (int j = 0; j < 2; ++j)
#pragma unroll
            for (int v = 0; v < 8; ++v) {
                const int row = bm + wr * 64 + i * 16 + v + 8 * hi;
                const int col = bn + wc * 32 + j * 16 + lo;
                store_elem(C, (size_t)row * N + col, acc[i][j][v]);
            }
}

// ---------- RoPE (in-place on bf16 projections) ----------
__global__ __launch_bounds__(256) void rope_kernel(unsigned short* __restrict__ X,
                                                   const float* __restrict__ fcos,
                                                   const float* __restrict__ fsin,
                                                   int nh) {
    const int total = BB * SS * nh * (HD / 2);
    int idx = blockIdx.x * blockDim.x + threadIdx.x;
    if (idx >= total) return;
    const int i = idx & 63;
    const int h = (idx >> 6) % nh;
    const int s = (idx / (64 * nh)) % SS;
    const int b = idx / (64 * nh * SS);
    unsigned short* base = X + ((size_t)(b * SS + s) * nh + h) * HD;
    const float c  = fcos[s * 64 + i];
    const float sn = fsin[s * 64 + i];
    const float x0 = bf2f(base[i]);
    const float x1 = bf2f(base[i + 64]);
    base[i]      = f2bf(x0 * c - x1 * sn);
    base[i + 64] = f2bf(x1 * c + x0 * sn);
}

// ---------- V transpose: (b,s,kvh,hd) -> (b,kvh,hd,S) so AV B-frags read 16B chunks ----------
__global__ __launch_bounds__(256) void vtrans_kernel(const unsigned short* __restrict__ V,
                                                     unsigned short* __restrict__ Vt) {
    const int total = BB * SS * KVH * HD;
    int idx = blockIdx.x * blockDim.x + threadIdx.x;
    if (idx >= total) return;
    const int hd  = idx & (HD - 1);
    const int kv  = (idx >> 7) & (KVH - 1);
    const int s   = (idx / (HD * KVH)) % SS;
    const int b   = idx / (HD * KVH * SS);
    Vt[(((size_t)(b * KVH + kv) * HD) + hd) * SS + s] = V[idx];
}

// ---------- Flash attention: per wave 16 q-rows, 32-key tiles, online softmax ----------
// Softmax kept in exp2 domain: scores scaled by (1/sqrt(d))*log2(e) so each
// probability is one hardware v_exp_f32.
__global__ __launch_bounds__(256) void attn_kernel(const unsigned short* __restrict__ Q,
                                                   const unsigned short* __restrict__ K,
                                                   const unsigned short* __restrict__ Vt,
                                                   unsigned short* __restrict__ O) {
    __shared__ unsigned short Plds[8][16 * 40];   // per-wave P staging tile

    const int b    = blockIdx.y / HH;
    const int h    = blockIdx.y % HH;
    const int kvh  = h / NREP;
    const int wave = threadIdx.x >> 5;
    const int lane = threadIdx.x & 31;
    const int hi   = lane >> 4, lo = lane & 15;
    const int q0   = blockIdx.x * 128 + wave * 16;

    // Cache Q A-fragments (16 rows x HD=128, 4 k-steps of 32)
    v16bf aQ[4];
    {
        const unsigned short* qrow = Q + (size_t)(b * SS + q0 + lo) * (HH * HD) + h * HD;
#pragma unroll
        for (int st = 0; st < 4; ++st) aQ[st] = glb_frag(qrow + st * 32 + hi * 8);
    }

    v8f acc[8];
#pragma unroll
    for (int j = 0; j < 8; ++j) acc[j] = (v8f)(0.0f);
    float mrow[8], lrow[8];
#pragma unroll
    for (int v = 0; v < 8; ++v) { mrow[v] = -3.0e38f; lrow[v] = 0.0f; }

    const float scale2 = 0.08838834764831845f * 1.4426950408889634f;  // (1/sqrt(128))*log2e
    const int nkt = (q0 + 16 + 31) >> 5;   // causal tile count for this wave

    for (int kt = 0; kt < nkt; ++kt) {
        const int kb = kt * 32;

        // ---- scores: two 16x16 tiles over 32 keys, chained over HD ----
        v8f sc[2];
        sc[0] = (v8f)(0.0f); sc[1] = (v8f)(0.0f);
#pragma unroll
        for (int t = 0; t < 2; ++t) {
            const unsigned short* krow =
                K + (size_t)(b * SS + kb + t * 16 + lo) * (KVH * HD) + kvh * HD;
            if (kt + 1 < nkt)                                     // global_prefetch_b8
                __builtin_prefetch(krow + (size_t)32 * (KVH * HD), 0, 1);
#pragma unroll
            for (int st = 0; st < 4; ++st) {
                v16bf bK = glb_frag(krow + st * 32 + hi * 8);
                sc[t] = wmma_bf16(aQ[st], bK, sc[t]);
            }
        }

        // ---- mask + scale (exp2 domain), online softmax (row = v + 8*hi) ----
        float nm[8];
#pragma unroll
        for (int v = 0; v < 8; ++v) {
            const int row = q0 + v + 8 * hi;
            float s0 = sc[0][v] * scale2;
            float s1 = sc[1][v] * scale2;
            if (kb + lo      > row) s0 = -3.0e38f;
            if (kb + 16 + lo > row) s1 = -3.0e38f;
            sc[0][v] = s0; sc[1][v] = s1;
            nm[v] = grp_max(fmaxf(fmaxf(s0, s1), mrow[v]));
        }
        float rfac[8], psum[8];
#pragma unroll
        for (int v = 0; v < 8; ++v) {
            rfac[v] = exp2f(mrow[v] - nm[v]);
            mrow[v] = nm[v];
            const float p0 = exp2f(sc[0][v] - nm[v]);
            const float p1 = exp2f(sc[1][v] - nm[v]);
            sc[0][v] = p0; sc[1][v] = p1;
            psum[v] = p0 + p1;
        }
#pragma unroll
        for (int v = 0; v < 8; ++v) {
            lrow[v] = lrow[v] * rfac[v] + grp_sum(psum[v]);
        }
#pragma unroll
        for (int j = 0; j < 8; ++j)
#pragma unroll
            for (int v = 0; v < 8; ++v) acc[j][v] *= rfac[v];

        // ---- relayout P (C-layout f32) -> A-fragment via per-wave LDS tile ----
        unsigned short* P = Plds[wave];
#pragma unroll
        for (int t = 0; t < 2; ++t)
#pragma unroll
            for (int v = 0; v < 8; ++v)
                P[(v + 8 * hi) * 40 + t * 16 + lo] = f2bf(sc[t][v]);
        asm volatile("s_wait_dscnt 0" ::: "memory");   // wave-local LDS RAW ordering
        v16bf aP = lds_frag(P, 0, lane);

        // ---- AV: 8 output tiles across HD; V^T gives contiguous key chunks ----
#pragma unroll
        for (int j = 0; j < 8; ++j) {
            const unsigned short* vrow =
                Vt + ((size_t)(b * KVH + kvh) * HD + j * 16 + lo) * SS + kb + hi * 8;
            v16bf bV = glb_frag(vrow);
            acc[j] = wmma_bf16(aP, bV, acc[j]);
        }
    }

    // ---- normalize and store bf16 attention output (b, s, h*HD) ----
#pragma unroll
    for (int v = 0; v < 8; ++v) lrow[v] = 1.0f / lrow[v];
#pragma unroll
    for (int j = 0; j < 8; ++j)
#pragma unroll
        for (int v = 0; v < 8; ++v) {
            const int row = q0 + v + 8 * hi;
            const int col = h * HD + j * 16 + lo;
            O[(size_t)(b * SS + row) * (HH * HD) + col] = f2bf(acc[j][v] * lrow[v]);
        }
}

// ---------- host-side launch ----------
static inline size_t alignUp(size_t x) { return (x + 255) & ~(size_t)255; }

extern "C" void kernel_launch(void* const* d_in, const int* in_sizes, int n_in,
                              void* d_out, int out_size, void* d_ws, size_t ws_size,
                              hipStream_t stream) {
    (void)in_sizes; (void)n_in; (void)out_size; (void)ws_size;
    const float* x    = (const float*)d_in[0];
    const float* fcos = (const float*)d_in[1];
    const float* fsin = (const float*)d_in[2];
    // d_in[3] = additive mask: causal handled analytically in-kernel
    const float* wq   = (const float*)d_in[4];
    const float* wk   = (const float*)d_in[5];
    const float* wv   = (const float*)d_in[6];
    const float* wo   = (const float*)d_in[7];
    float* out        = (float*)d_out;

    const int M = BB * SS;   // 4096 token rows
    char* ws = (char*)d_ws;
    size_t off = 0;
    auto carve = [&](size_t bytes) { char* p = ws + off; off += alignUp(bytes); return p; };

    unsigned short* xb  = (unsigned short*)carve((size_t)M * DD * 2);
    unsigned short* wqb = (unsigned short*)carve((size_t)DD * DD * 2);
    unsigned short* wkb = (unsigned short*)carve((size_t)KVH * HD * DD * 2);
    unsigned short* wvb = (unsigned short*)carve((size_t)KVH * HD * DD * 2);
    unsigned short* wob = (unsigned short*)carve((size_t)DD * DD * 2);
    unsigned short* Qb  = (unsigned short*)carve((size_t)M * HH * HD * 2);
    unsigned short* Kb  = (unsigned short*)carve((size_t)M * KVH * HD * 2);
    unsigned short* Vb  = (unsigned short*)carve((size_t)M * KVH * HD * 2);
    unsigned short* Vtb = (unsigned short*)carve((size_t)M * KVH * HD * 2);
    unsigned short* Ob  = (unsigned short*)carve((size_t)M * HH * HD * 2);

    // 1) convert inputs to bf16
    auto cvt = [&](const float* src, unsigned short* dst, size_t n) {
        int n4 = (int)(n / 4);
        int blocks = (n4 + 255) / 256;
        cvt_f32_bf16_kernel<<<blocks, 256, 0, stream>>>(src, dst, n4);
    };
    cvt(x,  xb,  (size_t)M * DD);
    cvt(wq, wqb, (size_t)DD * DD);
    cvt(wk, wkb, (size_t)KVH * HD * DD);
    cvt(wv, wvb, (size_t)KVH * HD * DD);
    cvt(wo, wob, (size_t)DD * DD);

    // 2) QKV projections (async-LDS double-buffered WMMA GEMMs, bf16 out)
    gemm_bf16_kernel<unsigned short><<<dim3(DD / 128, M / 128), 256, 0, stream>>>(
        xb, wqb, Qb, M, DD, DD);
    gemm_bf16_kernel<unsigned short><<<dim3((KVH * HD) / 128, M / 128), 256, 0, stream>>>(
        xb, wkb, Kb, M, KVH * HD, DD);
    gemm_bf16_kernel<unsigned short><<<dim3((KVH * HD) / 128, M / 128), 256, 0, stream>>>(
        xb, wvb, Vb, M, KVH * HD, DD);

    // 3) RoPE on Q and K
    {
        int nq = BB * SS * HH * (HD / 2);
        rope_kernel<<<(nq + 255) / 256, 256, 0, stream>>>(Qb, fcos, fsin, HH);
        int nk = BB * SS * KVH * (HD / 2);
        rope_kernel<<<(nk + 255) / 256, 256, 0, stream>>>(Kb, fcos, fsin, KVH);
    }

    // 4) transpose V for AV fragment loads
    {
        int nv = BB * SS * KVH * HD;
        vtrans_kernel<<<(nv + 255) / 256, 256, 0, stream>>>(Vb, Vtb);
    }

    // 5) flash attention (WMMA scores + WMMA AV, online softmax)
    attn_kernel<<<dim3(SS / 128, BB * HH), 256, 0, stream>>>(Qb, Kb, Vtb, Ob);

    // 6) output projection -> f32 result
    gemm_bf16_kernel<float><<<dim3(DD / 128, M / 128), 256, 0, stream>>>(
        Ob, wob, out, M, DD, DD);
}
